// MambaBlock_18262200943526
// MI455X (gfx1250) — matrix operand order repaired and checked by reading, not compiled, not run on hardware
//
#include <hip/hip_runtime.h>
#include <hip/hip_bf16.h>
#include <hip/hip_fp16.h>

// ---------------------------------------------------------------------------
// Mamba block for MI455X (gfx1250, wave32, WMMA + TDM).
// ---------------------------------------------------------------------------

typedef _Float16 half_t;
typedef __attribute__((ext_vector_type(16))) _Float16 v16h;
typedef __attribute__((ext_vector_type(8)))  _Float16 v8h;
typedef __attribute__((ext_vector_type(8)))  float    v8f;
typedef unsigned int v4u  __attribute__((ext_vector_type(4)));
typedef int          v4i_ __attribute__((ext_vector_type(4)));
typedef int          v8i_ __attribute__((ext_vector_type(8)));

#define NROWS  32768   // 8 sequences * 4096 tokens
#define LSEQ   4096
#define DMODEL 256
#define DINNER 512
#define DXZ    1024
#define DXDBL  144
#define DSTATE 64
#define NTOK   16384   // 4 * 64 * 64 (original tokens)

#ifndef __has_builtin
#define __has_builtin(x) 0
#endif
#if __has_builtin(__builtin_amdgcn_tensor_load_to_lds) && \
    __has_builtin(__builtin_amdgcn_s_wait_tensorcnt)
#define USE_TDM 1
#else
#define USE_TDM 0
#endif

// ---------------------------------------------------------------------------
// TDM: 2D f32 tile load Global -> LDS via Tensor Data Mover descriptor.
// Bit packing per CDNA5 ISA D# groups 0/1 (2D tensor; groups 2/3 unused).
// ---------------------------------------------------------------------------
#if USE_TDM
__device__ __forceinline__ void tdm_load_2d_f32(
    unsigned lds_off, const void* gptr,
    unsigned tile_d0, unsigned tile_d1,
    unsigned tensor_d0, unsigned tensor_d1,
    unsigned stride_elems)
{
    unsigned long long ga = (unsigned long long)(uintptr_t)gptr;
    v4u g0;
    g0[0] = 1u;                                           // count=1, user D#
    g0[1] = lds_off;                                      // lds_addr (bytes)
    g0[2] = (unsigned)(ga & 0xFFFFFFFFu);                 // global_addr[31:0]
    g0[3] = (unsigned)((ga >> 32) & 0x01FFFFFFu)          // global_addr[56:32]
            | (2u << 30);                                 // type = 2 (image)
    v8i_ g1;
    g1[0] = (int)(2u << 16);                              // data_size=2 -> 4B
    g1[1] = (int)((tensor_d0 & 0xFFFFu) << 16);           // tensor_dim0[15:0]
    g1[2] = (int)(((tensor_d0 >> 16) & 0xFFFFu)
            | ((tensor_d1 & 0xFFFFu) << 16));             // td0[31:16] | td1[15:0]
    g1[3] = (int)(((tensor_d1 >> 16) & 0xFFFFu)
            | ((tile_d0 & 0xFFFFu) << 16));               // td1[31:16] | tile_dim0
    g1[4] = (int)(tile_d1 & 0xFFFFu);                     // tile_dim1 | tile_dim2=0
    g1[5] = (int)stride_elems;                            // tensor_dim0_stride[31:0]
    g1[6] = 0;                                            // stride hi | dim1_stride lo
    g1[7] = 0;
    v4i_ z4 = {0, 0, 0, 0};
#if __clang_major__ >= 23
    v8i_ z8 = {0, 0, 0, 0, 0, 0, 0, 0};
    __builtin_amdgcn_tensor_load_to_lds(g0, g1, z4, z4, z8, 0);
#else
    __builtin_amdgcn_tensor_load_to_lds(g0, g1, z4, z4, 0);
#endif
}
#endif

// ---------------------------------------------------------------------------
// WMMA fragment loads (gfx1250 16x16x32 f16 layouts, wave32)
// ---------------------------------------------------------------------------
__device__ __forceinline__ v16h load_A_frag(const half_t* __restrict__ A, int lda,
                                            int m0, int k0, int lane) {
    // A: 16x32 (MxK). lane<16: row=lane, halves 0..7 -> K[k0..k0+7], 8..15 -> K[k0+16..k0+23]
    //                 lane>=16: row=lane-16, halves 0..7 -> K[k0+8..], 8..15 -> K[k0+24..]
    int lrow = lane & 15;
    int hs   = lane >> 4;
    const half_t* p = A + (size_t)(m0 + lrow) * lda + k0 + 8 * hs;
    union { v16h v; v8h h[2]; } u;
    u.h[0] = *(const v8h*)(p);
    u.h[1] = *(const v8h*)(p + 16);
    return u.v;
}

__device__ __forceinline__ v16h load_B_frag(const half_t* __restrict__ W, int ldw,
                                            int n0, int k0, int lane) {
    // B: 32x16 (KxN), B[k][n] = W[n][k]. lane<16: col=lane, halves 0..7 -> K[k0..k0+7], 8..15 -> K[k0+16..k0+23];
    // lane>=16: col=lane-16, halves 0..7 -> K[k0+8..], 8..15 -> K[k0+24..].
    int lcol = lane & 15;
    int hs   = lane >> 4;
    const half_t* p = W + (size_t)(n0 + lcol) * ldw + k0 + 8 * hs;
    union { v16h v; v8h h[2]; } u;
    u.h[0] = *(const v8h*)(p);
    u.h[1] = *(const v8h*)(p + 16);
    return u.v;
}

// ---------------------------------------------------------------------------
// Generic WMMA GEMM: out[M][N] = A[M][K] @ W[N][K]^T   (f16 in, f32 acc)
// mode 0: store f32; mode 1: store f16; mode 2: softplus(acc + bias[n]) f32
// Block = 128 threads = 4 waves; each wave owns one 16x16 tile.
// ---------------------------------------------------------------------------
__global__ __launch_bounds__(128) void gemm_wmma_kernel(
    const half_t* __restrict__ A, const half_t* __restrict__ W,
    int M, int N, int K,
    float* __restrict__ outF, half_t* __restrict__ outH,
    const float* __restrict__ bias, int mode)
{
    int lane = threadIdx.x & 31;
    int wave = threadIdx.x >> 5;
    int m0 = blockIdx.x * 16;
    int n0 = (blockIdx.y * 4 + wave) * 16;
    if (n0 >= N) return;

    v8f acc = {};
    for (int k0 = 0; k0 < K; k0 += 32) {
        v16h a = load_A_frag(A, K, m0, k0, lane);
        v16h b = load_B_frag(W, K, n0, k0, lane);
        acc = __builtin_amdgcn_wmma_f32_16x16x32_f16(
            false, a, false, b, (short)0, acc, false, false);
    }

    // C/D layout: lane<16 -> rows m0+0..7, lane>=16 -> rows m0+8..15; col = n0+(lane&15)
    int col   = n0 + (lane & 15);
    int rbase = m0 + ((lane >> 4) ? 8 : 0);
    if (mode == 0) {
#pragma unroll
        for (int v = 0; v < 8; ++v)
            outF[(size_t)(rbase + v) * N + col] = acc[v];
    } else if (mode == 1) {
#pragma unroll
        for (int v = 0; v < 8; ++v)
            outH[(size_t)(rbase + v) * N + col] = (half_t)acc[v];
    } else {
        float bv = bias[col];
#pragma unroll
        for (int v = 0; v < 8; ++v) {
            float x  = acc[v] + bv;
            float sp = (x > 20.0f) ? x : log1pf(__expf(x));
            outF[(size_t)(rbase + v) * N + col] = sp;
        }
    }
}

// ---------------------------------------------------------------------------
// Weight conversions
// ---------------------------------------------------------------------------
__global__ void f32_to_f16_kernel(const float* __restrict__ s, half_t* __restrict__ d, int n) {
    int i = blockIdx.x * blockDim.x + threadIdx.x;
    if (i < n) d[i] = (half_t)s[i];
}

__global__ void pack_wdt_kernel(const float* __restrict__ s, half_t* __restrict__ d) {
    int i = blockIdx.x * blockDim.x + threadIdx.x;   // 512*32
    if (i >= 512 * 32) return;
    int r = i >> 5, c = i & 31;
    d[i] = (c < 16) ? (half_t)s[r * 16 + c] : (half_t)0.0f;
}

__global__ void pack_dtr_kernel(const float* __restrict__ xdbl, half_t* __restrict__ dtr) {
    size_t i = (size_t)blockIdx.x * blockDim.x + threadIdx.x;   // NROWS*32
    if (i >= (size_t)NROWS * 32) return;
    size_t r = i >> 5; int c = (int)(i & 31);
    dtr[i] = (c < 16) ? (half_t)xdbl[r * DXDBL + c] : (half_t)0.0f;
}

// ---------------------------------------------------------------------------
// LayerNorm-1 + build both sequence views (row-major and col-major) in f16
// ---------------------------------------------------------------------------
__global__ __launch_bounds__(256) void ln1_build_xm_kernel(
    const float* __restrict__ x, const float* __restrict__ w,
    const float* __restrict__ b, half_t* __restrict__ xm)
{
    __shared__ float red[256];
    int bid = blockIdx.x;                 // b*4096 + i*64 + j
    int t   = threadIdx.x;
    int bb  = bid >> 12;
    int i   = (bid >> 6) & 63;
    int j   = bid & 63;

    float v = x[(size_t)bid * DMODEL + t];
    red[t] = v; __syncthreads();
    for (int s = 128; s > 0; s >>= 1) { if (t < s) red[t] += red[t + s]; __syncthreads(); }
    float mu = red[0] * (1.0f / 256.0f);
    __syncthreads();
    float d = v - mu;
    red[t] = d * d; __syncthreads();
    for (int s = 128; s > 0; s >>= 1) { if (t < s) red[t] += red[t + s]; __syncthreads(); }
    float var = red[0] * (1.0f / 256.0f);

    half_t hv = (half_t)(d * rsqrtf(var + 1e-6f) * w[t] + b[t]);
    xm[((size_t)bb * LSEQ + i * 64 + j) * DMODEL + t]       = hv;  // x1 view
    xm[((size_t)(4 + bb) * LSEQ + j * 64 + i) * DMODEL + t] = hv;  // x2 view
}

// ---------------------------------------------------------------------------
// Depthwise causal conv (k=4) + bias + SiLU, on xc = xz[:, :512]
// ---------------------------------------------------------------------------
__global__ __launch_bounds__(256) void conv_silu_kernel(
    const half_t* __restrict__ xz, const float* __restrict__ cw,
    const float* __restrict__ cb, half_t* __restrict__ xc)
{
    size_t idx = (size_t)blockIdx.x * blockDim.x + threadIdx.x;   // NROWS*512
    if (idx >= (size_t)NROWS * DINNER) return;
    int    d = (int)(idx & 511);
    size_t r = idx >> 9;
    int    t = (int)(r & (LSEQ - 1));
    float acc = cb[d];
#pragma unroll
    for (int k = 0; k < 4; ++k) {
        int tt = t - 3 + k;
        if (tt >= 0)
            acc += (float)xz[(r - 3 + k) * DXZ + d] * cw[d * 4 + k];
    }
    float sig = 1.0f / (1.0f + __expf(-acc));
    xc[idx] = (half_t)(acc * sig);
}

// ---------------------------------------------------------------------------
// Selective scan, fused with +u*D and *silu(z), f16 output.
// grid (64, 8): blockIdx.x = channel group of 8, blockIdx.y = sequence.
// 256 threads = 8 waves; one wave per channel d; lane holds states s, s+32.
// B/C chunks staged via Tensor Data Mover (wave 0 issues DMA, waits
// TENSORcnt); dt/u/z staged by the other lanes in parallel.
// ---------------------------------------------------------------------------
__global__ __launch_bounds__(256) void scan_kernel(
    const half_t* __restrict__ u_h,   // [NROWS][512]
    const float*  __restrict__ dt,    // [NROWS][512]
    const float*  __restrict__ xdbl,  // [NROWS][144] (dt_r|B|C)
    const half_t* __restrict__ xz_h,  // [NROWS][1024] (z at 512+)
    const float*  __restrict__ A_log, // [512][64]
    const float*  __restrict__ Dp,    // [512]
    half_t* __restrict__ ym)          // [NROWS][512]
{
    __shared__ float  Bs[64][64];
    __shared__ float  Cs[64][64];
    __shared__ float  dts[8][64];
    __shared__ float  us[8][64];
    __shared__ float  zs[8][64];
    __shared__ half_t ys[8][64];

    int tid  = threadIdx.x;
    int lane = tid & 31;
    int wave = tid >> 5;
    int bb   = blockIdx.y;
    int dg   = blockIdx.x;
    int d    = dg * 8 + wave;

    float A0 = -__expf(A_log[d * DSTATE + lane]);
    float A1 = -__expf(A_log[d * DSTATE + lane + 32]);
    float Dd = Dp[d];
    float h0 = 0.0f, h1 = 0.0f;
    size_t rowbase = (size_t)bb * LSEQ;

#if USE_TDM
    unsigned bs_off = (unsigned)(uintptr_t)&Bs[0][0];
    unsigned cs_off = (unsigned)(uintptr_t)&Cs[0][0];
#endif

    for (int t0 = 0; t0 < LSEQ; t0 += 64) {
#if USE_TDM
        if (wave == 0) {
            const float* src = xdbl + (rowbase + t0) * DXDBL;
            // 64x64 f32 tile, row stride 144 floats
            tdm_load_2d_f32(bs_off, src + 16, 64, 64, 64, 64, DXDBL);
            tdm_load_2d_f32(cs_off, src + 80, 64, 64, 64, 64, DXDBL);
        }
#else
        for (int e = tid; e < 4096; e += 256) {
            int i = e >> 6, j = e & 63;
            const float* src = xdbl + (rowbase + t0 + i) * DXDBL;
            Bs[i][j] = src[16 + j];
            Cs[i][j] = src[80 + j];
        }
#endif
        for (int e = tid; e < 512; e += 256) {
            int w = e >> 6, tt = e & 63;
            size_t row = rowbase + t0 + tt;
            int    dd  = dg * 8 + w;
            dts[w][tt] = dt[row * DINNER + dd];
            us[w][tt]  = (float)u_h[row * DINNER + dd];
            zs[w][tt]  = (float)xz_h[row * DXZ + DINNER + dd];
        }
#if USE_TDM
        if (wave == 0) __builtin_amdgcn_s_wait_tensorcnt(0);
#endif
        __syncthreads();

        for (int tt = 0; tt < 64; ++tt) {
            float dtv = dts[wave][tt];
            float uv  = us[wave][tt];
            float du  = dtv * uv;
            float b0 = Bs[tt][lane],      c0 = Cs[tt][lane];
            float b1 = Bs[tt][lane + 32], c1 = Cs[tt][lane + 32];
            h0 = __expf(dtv * A0) * h0 + du * b0;
            h1 = __expf(dtv * A1) * h1 + du * b1;
            float sum = h0 * c0 + h1 * c1;
#pragma unroll
            for (int off = 16; off > 0; off >>= 1)
                sum += __shfl_xor(sum, off, 32);
            if (lane == 0) {
                float y   = sum + uv * Dd;
                float z   = zs[wave][tt];
                float sil = z / (1.0f + __expf(-z));
                ys[wave][tt] = (half_t)(y * sil);
            }
        }
        __syncthreads();

        for (int e = tid; e < 512; e += 256) {
            int w = e & 7, tt = e >> 3;
            ym[(rowbase + t0 + tt) * DINNER + dg * 8 + w] = ys[w][tt];
        }
        __syncthreads();
    }
}

// ---------------------------------------------------------------------------
// LayerNorm-2 on both views + residual combine: out = x + ln(y1) + ln(y2)
// ---------------------------------------------------------------------------
__global__ __launch_bounds__(256) void final_combine_kernel(
    const float* __restrict__ yproj,  // [NROWS][256]
    const float* __restrict__ x,
    const float* __restrict__ w2, const float* __restrict__ b2,
    float* __restrict__ out)
{
    __shared__ float red1[256];
    __shared__ float red2[256];
    int bid = blockIdx.x;
    int t   = threadIdx.x;
    int bb  = bid >> 12;
    int i   = (bid >> 6) & 63;
    int j   = bid & 63;

    float v1 = yproj[((size_t)bb * LSEQ + i * 64 + j) * DMODEL + t];
    float v2 = yproj[((size_t)(4 + bb) * LSEQ + j * 64 + i) * DMODEL + t];

    red1[t] = v1; red2[t] = v2; __syncthreads();
    for (int s = 128; s > 0; s >>= 1) {
        if (t < s) { red1[t] += red1[t + s]; red2[t] += red2[t + s]; }
        __syncthreads();
    }
    float mu1 = red1[0] * (1.0f / 256.0f);
    float mu2 = red2[0] * (1.0f / 256.0f);
    __syncthreads();
    float d1 = v1 - mu1, d2 = v2 - mu2;
    red1[t] = d1 * d1; red2[t] = d2 * d2; __syncthreads();
    for (int s = 128; s > 0; s >>= 1) {
        if (t < s) { red1[t] += red1[t + s]; red2[t] += red2[t + s]; }
        __syncthreads();
    }
    float var1 = red1[0] * (1.0f / 256.0f);
    float var2 = red2[0] * (1.0f / 256.0f);

    float wv = w2[t], bv = b2[t];
    float n1 = d1 * rsqrtf(var1 + 1e-6f) * wv + bv;
    float n2 = d2 * rsqrtf(var2 + 1e-6f) * wv + bv;
    out[(size_t)bid * DMODEL + t] = x[(size_t)bid * DMODEL + t] + n1 + n2;
}

// ---------------------------------------------------------------------------
// Launch
// ---------------------------------------------------------------------------
static constexpr size_t align256(size_t x) { return (x + 255) & ~size_t(255); }

extern "C" void kernel_launch(void* const* d_in, const int* in_sizes, int n_in,
                              void* d_out, int out_size, void* d_ws, size_t ws_size,
                              hipStream_t stream) {
    const float* x     = (const float*)d_in[0];
    const float* ln1w  = (const float*)d_in[1];
    const float* ln1b  = (const float*)d_in[2];
    const float* ln2w  = (const float*)d_in[3];
    const float* ln2b  = (const float*)d_in[4];
    const float* W_in  = (const float*)d_in[5];
    const float* convw = (const float*)d_in[6];
    const float* convb = (const float*)d_in[7];
    const float* W_x   = (const float*)d_in[8];
    const float* W_dt  = (const float*)d_in[9];
    const float* b_dt  = (const float*)d_in[10];
    const float* A_log = (const float*)d_in[11];
    const float* Dp    = (const float*)d_in[12];
    const float* W_out = (const float*)d_in[13];
    float* out = (float*)d_out;

    char* ws = (char*)d_ws;
    size_t off = 0;
    half_t* xm_h   = (half_t*)(ws + off); off = align256(off + (size_t)NROWS * DMODEL * 2);
    half_t* xz_h   = (half_t*)(ws + off); off = align256(off + (size_t)NROWS * DXZ   * 2);
    half_t* xc_h   = (half_t*)(ws + off); off = align256(off + (size_t)NROWS * DINNER* 2);
    float*  xdbl   = (float*) (ws + off); off = align256(off + (size_t)NROWS * DXDBL * 4);
    half_t* dtr_h  = (half_t*)(ws + off); off = align256(off + (size_t)NROWS * 32    * 2);
    float*  dtbuf  = (float*) (ws + off); off = align256(off + (size_t)NROWS * DINNER* 4);
    half_t* ym_h   = (half_t*)(ws + off); off = align256(off + (size_t)NROWS * DINNER* 2);
    float*  yproj  = (float*) (ws + off); off = align256(off + (size_t)NROWS * DMODEL* 4);
    half_t* win_h  = (half_t*)(ws + off); off = align256(off + (size_t)DXZ   * DMODEL* 2);
    half_t* wx_h   = (half_t*)(ws + off); off = align256(off + (size_t)DXDBL * DINNER* 2);
    half_t* wdt_h  = (half_t*)(ws + off); off = align256(off + (size_t)DINNER* 32    * 2);
    half_t* wout_h = (half_t*)(ws + off); off = align256(off + (size_t)DMODEL* DINNER* 2);
    (void)ws_size; (void)in_sizes; (void)n_in; (void)out_size;

    // 0) weights -> f16
    f32_to_f16_kernel<<<(DXZ * DMODEL + 255) / 256, 256, 0, stream>>>(W_in, win_h, DXZ * DMODEL);
    f32_to_f16_kernel<<<(DXDBL * DINNER + 255) / 256, 256, 0, stream>>>(W_x, wx_h, DXDBL * DINNER);
    f32_to_f16_kernel<<<(DMODEL * DINNER + 255) / 256, 256, 0, stream>>>(W_out, wout_h, DMODEL * DINNER);
    pack_wdt_kernel<<<(512 * 32 + 255) / 256, 256, 0, stream>>>(W_dt, wdt_h);

    // 1) LN1 + build both sequence views
    ln1_build_xm_kernel<<<NTOK, 256, 0, stream>>>(x, ln1w, ln1b, xm_h);

    // 2) xz = xm @ W_in^T  (M=32768, N=1024, K=256) -> f16
    gemm_wmma_kernel<<<dim3(NROWS / 16, DXZ / 64), 128, 0, stream>>>(
        xm_h, win_h, NROWS, DXZ, DMODEL, nullptr, xz_h, nullptr, 1);

    // 3) depthwise conv + SiLU on xc half
    conv_silu_kernel<<<(unsigned)(((size_t)NROWS * DINNER + 255) / 256), 256, 0, stream>>>(
        xz_h, convw, convb, xc_h);

    // 4) x_dbl = xc @ W_x^T  (M=32768, N=144, K=512) -> f32
    gemm_wmma_kernel<<<dim3(NROWS / 16, (DXDBL / 16 + 3) / 4), 128, 0, stream>>>(
        xc_h, wx_h, NROWS, DXDBL, DINNER, xdbl, nullptr, nullptr, 0);

    // 5) pack dt_r (pad K 16->32)
    pack_dtr_kernel<<<(unsigned)(((size_t)NROWS * 32 + 255) / 256), 256, 0, stream>>>(xdbl, dtr_h);

    // 6) dt = softplus(dt_r @ W_dt^T + b_dt)  (M=32768, N=512, K=32) -> f32
    gemm_wmma_kernel<<<dim3(NROWS / 16, DINNER / 64), 128, 0, stream>>>(
        dtr_h, wdt_h, NROWS, DINNER, 32, dtbuf, nullptr, b_dt, 2);

    // 7) selective scan + gate -> ym (f16)
    scan_kernel<<<dim3(64, 8), 256, 0, stream>>>(
        xc_h, dtbuf, xdbl, xz_h, A_log, Dp, ym_h);

    // 8) yproj = ym @ W_out^T  (M=32768, N=256, K=512) -> f32
    gemm_wmma_kernel<<<dim3(NROWS / 16, DMODEL / 64), 128, 0, stream>>>(
        ym_h, wout_h, NROWS, DMODEL, DINNER, yproj, nullptr, nullptr, 0);

    // 9) LN2 + residual combine of both views
    final_combine_kernel<<<NTOK, 256, 0, stream>>>(yproj, x, ln2w, ln2b, out);
}
